// BlockRecurrentAttention_19189913879041
// MI455X (gfx1250) — matrix operand order repaired and checked by reading, not compiled, not run on hardware
//
#include <hip/hip_runtime.h>

// ---------------------------------------------------------------------------
// Causal MHA block for MI455X (gfx1250, wave32, WMMA 16x16x32 f16, f32 accum)
// Async-to-LDS (ASYNCcnt) staging with software pipelining (ping-pong tiles).
// ---------------------------------------------------------------------------
typedef __attribute__((ext_vector_type(16))) _Float16 v16h;
typedef __attribute__((ext_vector_type(8)))  float    v8f;

static constexpr int BATCH = 2;
static constexpr int SEQ   = 2048;
static constexpr int DIM   = 1024;
static constexpr int HEADS = 16;
static constexpr int DH    = 64;
static constexpr int ROTD  = 32;

// A-fragment K index used with v_wmma_f32_16x16x32_f16 (contiguous order, same as the B loads):
// lanes 0-15 hold K in {0..15}, lanes 16-31 hold K in {16..31}
__device__ __forceinline__ int a_k_of_i(int i, int khalf) {
  return i + (khalf << 4);
}

__device__ __forceinline__ v8f wmma_f16(const v16h& a, const v16h& b, const v8f& c) {
  return __builtin_amdgcn_wmma_f32_16x16x32_f16(false, a, false, b, (short)0, c,
                                                false, false);
}

// CDNA5 async copy: 16 bytes global -> LDS, tracked by ASYNCcnt (no VGPR data).
__device__ __forceinline__ void async_b128(void* lds, const void* gptr) {
  unsigned lds_off = (unsigned)(unsigned long long)lds;   // low 32b = LDS offset
  asm volatile("global_load_async_to_lds_b128 %0, %1, off"
               :: "v"(lds_off), "v"(gptr) : "memory");
}
// Async loads complete in order: waiting <=4 forces all ops older than the
// 4 most recently issued (the prefetched next tile) to be complete.
__device__ __forceinline__ void wait_async0() {
  asm volatile("s_wait_asynccnt 0x0" ::: "memory");
}
__device__ __forceinline__ void wait_async4() {
  asm volatile("s_wait_asynccnt 0x4" ::: "memory");
}

// ---------------------------------------------------------------------------
// Kernel 1: fused QKV projection + RoPE.  Tile 128x64, 4 waves, each wave
// owns a 32x64 strip (2 A-frags x 4 B-frags -> 8 WMMAs per k-step).
// The 64-wide column tile == one head, so the RoPE pair (d, d+16) lives in
// accumulators 0/1 of the SAME lane.  Q,K -> [b,h,n,d] f16 ; V -> [b,h,d,n].
// ---------------------------------------------------------------------------
__global__ __launch_bounds__(128) void qkv_proj_rope_kernel(
    const float* __restrict__ x, const float* __restrict__ rot,
    const float* __restrict__ Wq, const float* __restrict__ Wk,
    const float* __restrict__ Wv,
    _Float16* __restrict__ Qh, _Float16* __restrict__ Kh,
    _Float16* __restrict__ Vt)
{
  __shared__ _Float16 As[128][32];   // [m][k]
  __shared__ _Float16 Bs[64][32];    // [n][k]

  const int tid  = threadIdx.x;
  const int wave = tid >> 5;
  const int lane = tid & 31;
  const int lcol = lane & 15;
  const int khalf = lane >> 4;

  const int rowBase = blockIdx.x * 128;   // over B*SEQ = 4096
  const int head    = blockIdx.y;
  const int colBase = head * 64;
  const int z       = blockIdx.z;         // 0=Q 1=K 2=V
  const float* W = (z == 0) ? Wq : (z == 1) ? Wk : Wv;

  v8f acc[2][4] = {};

  for (int kt = 0; kt < DIM; kt += 32) {
    if (kt + 32 < DIM)
      __builtin_prefetch(&W[(size_t)(kt + 32) * DIM + colBase + lane], 0, 0);

    // stage A (f32 -> f16): one 32-wide row per thread, vector global loads
    {
      const float4* src = (const float4*)(x + (size_t)(rowBase + tid) * DIM + kt);
      #pragma unroll
      for (int i = 0; i < 8; ++i) {
        float4 f = src[i];
        As[tid][i * 4 + 0] = (_Float16)f.x;
        As[tid][i * 4 + 1] = (_Float16)f.y;
        As[tid][i * 4 + 2] = (_Float16)f.z;
        As[tid][i * 4 + 3] = (_Float16)f.w;
      }
    }
    // stage B: W[k][n] -> Bs[n][k] (f32 -> f16)
    #pragma unroll
    for (int i = 0; i < 16; ++i) {
      int flat = tid * 16 + i;
      int n = flat >> 5, k = flat & 31;
      Bs[n][k] = (_Float16)W[(size_t)(kt + k) * DIM + colBase + n];
    }
    __syncthreads();

    // fragments: batch all loads, then the WMMA burst
    v16h a0, a1, bfr[4];
    const int mrow = wave * 32 + lcol;
    #pragma unroll
    for (int i = 0; i < 16; ++i) {
      a0[i] = As[mrow][a_k_of_i(i, khalf)];
      a1[i] = As[mrow + 16][a_k_of_i(i, khalf)];
    }
    #pragma unroll
    for (int s = 0; s < 4; ++s) {
      const int ncol = s * 16 + lcol;
      #pragma unroll
      for (int i = 0; i < 16; ++i) bfr[s][i] = Bs[ncol][khalf * 16 + i];
    }
    #pragma unroll
    for (int s = 0; s < 4; ++s) acc[0][s] = wmma_f16(a0, bfr[s], acc[0][s]);
    #pragma unroll
    for (int s = 0; s < 4; ++s) acc[1][s] = wmma_f16(a1, bfr[s], acc[1][s]);
    __syncthreads();
  }

  // epilogue: RoPE on d in [0,32), scale Q, f16 stores
  const int rofs = khalf * 8;
  #pragma unroll
  for (int p = 0; p < 2; ++p) {
    #pragma unroll
    for (int r = 0; r < 8; ++r) {
      const int grow  = rowBase + wave * 32 + p * 16 + r + rofs;
      const int b_idx = grow >> 11;
      const int npos  = grow & (SEQ - 1);
      float t0 = acc[p][0][r], t1 = acc[p][1][r];
      float t2 = acc[p][2][r], t3 = acc[p][3][r];

      const float f0 = rot[(size_t)npos * ROTD + lcol];
      const float f1 = rot[(size_t)npos * ROTD + 16 + lcol];
      float o0 = t0 * __cosf(f0) - t1 * __sinf(f0);
      float o1 = t1 * __cosf(f1) + t0 * __sinf(f1);
      float o2 = t2, o3 = t3;
      if (z == 0) { o0 *= 0.125f; o1 *= 0.125f; o2 *= 0.125f; o3 *= 0.125f; }

      const size_t bh = (size_t)(b_idx * HEADS + head);
      if (z == 2) {
        _Float16* dst = Vt + bh * DH * SEQ;           // [d][n]
        dst[(size_t)(lcol)      * SEQ + npos] = (_Float16)o0;
        dst[(size_t)(16 + lcol) * SEQ + npos] = (_Float16)o1;
        dst[(size_t)(32 + lcol) * SEQ + npos] = (_Float16)o2;
        dst[(size_t)(48 + lcol) * SEQ + npos] = (_Float16)o3;
      } else {
        _Float16* dst = ((z == 0) ? Qh : Kh) + (bh * SEQ + npos) * DH;
        dst[lcol]      = (_Float16)o0;
        dst[16 + lcol] = (_Float16)o1;
        dst[32 + lcol] = (_Float16)o2;
        dst[48 + lcol] = (_Float16)o3;
      }
    }
  }
}

// ---------------------------------------------------------------------------
// Kernel 2: causal flash attention.  One (b,h) x 128 queries per block,
// 8 waves x 16 query rows.  K/V tiles double-buffered: the next tile's
// async-to-LDS copies are in flight while the current tile is computed.
// ---------------------------------------------------------------------------
__global__ __launch_bounds__(256) void flash_attn_kernel(
    const _Float16* __restrict__ Qh, const _Float16* __restrict__ Kh,
    const _Float16* __restrict__ Vt, _Float16* __restrict__ Oh)
{
  __shared__ _Float16 Qs[128][64];       // [m][d]        16 KB
  __shared__ _Float16 Ks[2][64][64];     // [buf][key][d] 16 KB
  __shared__ _Float16 Vs[2][64][64];     // [buf][d][key] 16 KB
  __shared__ _Float16 Ps[8][16][64];     // per-wave P    16 KB

  const int tid  = threadIdx.x;
  const int wave = tid >> 5;
  const int lane = tid & 31;
  const int lcol = lane & 15;
  const int khalf = lane >> 4;
  const int rofs  = khalf * 8;

  const int qt    = blockIdx.x;
  const int bh    = blockIdx.y;
  const int qbase = qt * 128;
  const _Float16* Qp = Qh + ((size_t)bh * SEQ + qbase) * DH;
  const _Float16* Kp = Kh + (size_t)bh * SEQ * DH;
  const _Float16* Vp = Vt + (size_t)bh * DH * SEQ;

  // per-thread staging coordinates (2 K-chunks + 2 V-chunks per tile = 4 async)
  const int cc0 = tid * 2, r00 = cc0 >> 3, c80 = (cc0 & 7) * 8;
  const int cc1 = cc0 + 1, r01 = cc1 >> 3, c81 = (cc1 & 7) * 8;

  // async-stage Q tile: 128x64 f16 = 1024 b128 chunks, 4 per thread
  #pragma unroll
  for (int i = 0; i < 4; ++i) {
    const int cc = tid * 4 + i;              // chunk = 8 f16
    const int m = cc >> 3, c8 = (cc & 7) * 8;
    async_b128(&Qs[m][c8], Qp + (size_t)m * DH + c8);
  }
  // prime the pipeline: tile 0 into buffer 0
  async_b128(&Ks[0][r00][c80], Kp + (size_t)r00 * DH + c80);
  async_b128(&Ks[0][r01][c81], Kp + (size_t)r01 * DH + c81);
  async_b128(&Vs[0][r00][c80], Vp + (size_t)r00 * SEQ + c80);
  async_b128(&Vs[0][r01][c81], Vp + (size_t)r01 * SEQ + c81);

  v8f o[4] = {};
  float mstate[8], lstate[8];
  #pragma unroll
  for (int r = 0; r < 8; ++r) { mstate[r] = -3.0e38f; lstate[r] = 0.f; }

  const int jtmax = 2 * qt + 1;              // causal: keys <= qbase+127
  for (int jt = 0; jt <= jtmax; ++jt) {
    const int jbase = jt * 64;
    const int cur = jt & 1;
    if (jt < jtmax) {                        // prefetch next tile into other buf
      const int nb = jbase + 64;
      async_b128(&Ks[cur ^ 1][r00][c80], Kp + (size_t)(nb + r00) * DH + c80);
      async_b128(&Ks[cur ^ 1][r01][c81], Kp + (size_t)(nb + r01) * DH + c81);
      async_b128(&Vs[cur ^ 1][r00][c80], Vp + (size_t)r00 * SEQ + nb + c80);
      async_b128(&Vs[cur ^ 1][r01][c81], Vp + (size_t)r01 * SEQ + nb + c81);
      wait_async4();   // only the 4 just-issued may remain outstanding
    } else {
      wait_async0();
    }
    __syncthreads();   // tile jt (and Q on jt==0) visible to all waves

    // ---- S = Q K^T (K-dim = head dim, 2 steps of 32) ----
    v8f s[4] = {};
    #pragma unroll
    for (int ks = 0; ks < 2; ++ks) {
      v16h a, bfr[4];
      const int mrow = wave * 16 + lcol;
      #pragma unroll
      for (int i = 0; i < 16; ++i) a[i] = Qs[mrow][ks * 32 + a_k_of_i(i, khalf)];
      #pragma unroll
      for (int sub = 0; sub < 4; ++sub) {
        const int ncol = sub * 16 + lcol;
        #pragma unroll
        for (int i = 0; i < 16; ++i)
          bfr[sub][i] = Ks[cur][ncol][ks * 32 + khalf * 16 + i];
      }
      #pragma unroll
      for (int sub = 0; sub < 4; ++sub) s[sub] = wmma_f16(a, bfr[sub], s[sub]);
    }

    // ---- causal mask + online softmax ----
    #pragma unroll
    for (int r = 0; r < 8; ++r) {
      const int qrow = qbase + wave * 16 + r + rofs;
      float sv[4]; float mx = -3.0e38f;
      #pragma unroll
      for (int sub = 0; sub < 4; ++sub) {
        const int kcol = jbase + sub * 16 + lcol;
        float v = s[sub][r];
        if (kcol > qrow) v = -3.0e38f;
        sv[sub] = v; mx = fmaxf(mx, v);
      }
      #pragma unroll
      for (int off = 8; off >= 1; off >>= 1)
        mx = fmaxf(mx, __shfl_xor(mx, off, 32));
      const float newm = fmaxf(mstate[r], mx);
      const float corr = __expf(mstate[r] - newm);
      mstate[r] = newm;
      float rsum = 0.f;
      #pragma unroll
      for (int sub = 0; sub < 4; ++sub) {
        const float p = __expf(sv[sub] - newm);
        rsum += p;
        Ps[wave][r + rofs][sub * 16 + lcol] = (_Float16)p;
      }
      #pragma unroll
      for (int off = 8; off >= 1; off >>= 1)
        rsum += __shfl_xor(rsum, off, 32);
      lstate[r] = lstate[r] * corr + rsum;
      #pragma unroll
      for (int sub = 0; sub < 4; ++sub) o[sub][r] *= corr;
    }

    // ---- O += P V (K-dim = keys); per-wave LDS region, DScnt orders it ----
    #pragma unroll
    for (int ks = 0; ks < 2; ++ks) {
      v16h a, bfr[4];
      #pragma unroll
      for (int i = 0; i < 16; ++i)
        a[i] = Ps[wave][lcol][ks * 32 + a_k_of_i(i, khalf)];
      #pragma unroll
      for (int sub = 0; sub < 4; ++sub) {
        const int dcol = sub * 16 + lcol;
        #pragma unroll
        for (int i = 0; i < 16; ++i)
          bfr[sub][i] = Vs[cur][dcol][ks * 32 + khalf * 16 + i];
      }
      #pragma unroll
      for (int sub = 0; sub < 4; ++sub) o[sub] = wmma_f16(a, bfr[sub], o[sub]);
    }
    __syncthreads();   // everyone done with buf[cur] before it is refilled
  }

  // finalize: O /= l, write [b][n][h*64+d] f16
  const int b_idx = bh >> 4, h = bh & 15;
  #pragma unroll
  for (int r = 0; r < 8; ++r) {
    const int npos = qbase + wave * 16 + r + rofs;
    const float inv = 1.f / lstate[r];
    _Float16* dst = Oh + ((size_t)b_idx * SEQ + npos) * DIM + h * DH;
    #pragma unroll
    for (int sub = 0; sub < 4; ++sub)
      dst[sub * 16 + lcol] = (_Float16)(o[sub][r] * inv);
  }
}

// ---------------------------------------------------------------------------
// Kernel 3: output projection  out = Oh(4096x1024 f16) @ Wout -> f32.
// Tile 128x64; A tile async-staged and double-buffered across the k-loop.
// ---------------------------------------------------------------------------
__global__ __launch_bounds__(128) void out_proj_kernel(
    const _Float16* __restrict__ A, const float* __restrict__ W,
    float* __restrict__ C)
{
  __shared__ _Float16 As[2][128][32];    // ping-pong, 16 KB
  __shared__ _Float16 Bs[64][32];

  const int tid  = threadIdx.x;
  const int wave = tid >> 5;
  const int lane = tid & 31;
  const int lcol = lane & 15;
  const int khalf = lane >> 4;

  const int rowBase = blockIdx.x * 128;
  const int colBase = blockIdx.y * 64;

  // per-thread staging coordinates: 128x32 f16 = 512 chunks, 4 per thread
  const int m0 = (tid * 4) >> 2;                 // == tid
  // each thread owns one 32-wide A row = 4 b128 chunks
  const _Float16* Arow = A + (size_t)(rowBase + m0) * DIM;

  // prime: kt = 0 into buffer 0
  #pragma unroll
  for (int i = 0; i < 4; ++i)
    async_b128(&As[0][m0][i * 8], Arow + i * 8);

  v8f acc[2][4] = {};
  for (int kt = 0; kt < DIM; kt += 32) {
    const int cur = (kt >> 5) & 1;
    const bool has_next = (kt + 32 < DIM);
    if (has_next) {
      __builtin_prefetch(&W[(size_t)(kt + 32) * DIM + colBase + lane], 0, 0);
      #pragma unroll
      for (int i = 0; i < 4; ++i)
        async_b128(&As[cur ^ 1][m0][i * 8], Arow + kt + 32 + i * 8);
    }
    // stage B with f32 -> f16 conversion (VALU path)
    #pragma unroll
    for (int i = 0; i < 16; ++i) {
      int flat = tid * 16 + i;
      int n = flat >> 5, k = flat & 31;
      Bs[n][k] = (_Float16)W[(size_t)(kt + k) * DIM + colBase + n];
    }
    if (has_next) wait_async4(); else wait_async0();
    __syncthreads();

    v16h a0, a1, bfr[4];
    const int mrow = wave * 32 + lcol;
    #pragma unroll
    for (int i = 0; i < 16; ++i) {
      a0[i] = As[cur][mrow][a_k_of_i(i, khalf)];
      a1[i] = As[cur][mrow + 16][a_k_of_i(i, khalf)];
    }
    #pragma unroll
    for (int s = 0; s < 4; ++s) {
      const int ncol = s * 16 + lcol;
      #pragma unroll
      for (int i = 0; i < 16; ++i) bfr[s][i] = Bs[ncol][khalf * 16 + i];
    }
    #pragma unroll
    for (int s = 0; s < 4; ++s) acc[0][s] = wmma_f16(a0, bfr[s], acc[0][s]);
    #pragma unroll
    for (int s = 0; s < 4; ++s) acc[1][s] = wmma_f16(a1, bfr[s], acc[1][s]);
    __syncthreads();
  }

  const int rofs = khalf * 8;
  #pragma unroll
  for (int p = 0; p < 2; ++p) {
    #pragma unroll
    for (int r = 0; r < 8; ++r) {
      const int grow = rowBase + wave * 32 + p * 16 + r + rofs;
      #pragma unroll
      for (int s = 0; s < 4; ++s)
        C[(size_t)grow * DIM + colBase + s * 16 + lcol] = acc[p][s][r];
    }
  }
}

// ---------------------------------------------------------------------------
extern "C" void kernel_launch(void* const* d_in, const int* in_sizes, int n_in,
                              void* d_out, int out_size, void* d_ws, size_t ws_size,
                              hipStream_t stream) {
  const float* x    = (const float*)d_in[0];
  const float* rot  = (const float*)d_in[1];
  const float* Wq   = (const float*)d_in[2];
  const float* Wk   = (const float*)d_in[3];
  const float* Wv   = (const float*)d_in[4];
  const float* Wout = (const float*)d_in[5];
  float* out = (float*)d_out;

  const size_t per = (size_t)BATCH * HEADS * SEQ * DH;   // 4,194,304 f16 elems
  _Float16* Qh = (_Float16*)d_ws;
  _Float16* Kh = Qh + per;
  _Float16* Vt = Kh + per;
  _Float16* Oh = Vt + per;

  dim3 g1(BATCH * SEQ / 128, DIM / 64, 3);
  qkv_proj_rope_kernel<<<g1, 128, 0, stream>>>(x, rot, Wq, Wk, Wv, Qh, Kh, Vt);

  dim3 g2(SEQ / 128, BATCH * HEADS);
  flash_attn_kernel<<<g2, 256, 0, stream>>>(Qh, Kh, Vt, Oh);

  dim3 g3(BATCH * SEQ / 128, DIM / 64);
  out_proj_kernel<<<g3, 128, 0, stream>>>(Oh, Wout, out);
}
